// Kandinsky5Attention_6880537608327
// MI455X (gfx1250) — matrix operand order repaired and checked by reading, not compiled, not run on hardware
//
#include <hip/hip_runtime.h>

// ---------------------------------------------------------------------------
// Kandinsky5 Nabla block-sparse attention for MI455X (gfx1250, wave32, WMMA)
// All matmuls in bf16 via v_wmma_f32_16x16x32_bf16, f32 accumulate.
// GEMM tiles staged with the Tensor Data Mover (tensor_load_to_lds) when the
// builtin is available; TDM LDS-padding reproduces the 48-elem padded stride.
// ---------------------------------------------------------------------------

namespace k5 {
constexpr int S    = 2048;
constexpr int C    = 2048;
constexpr int NH   = 16;
constexpr int HD   = 128;
constexpr int BLK  = 64;
constexpr int S1   = 32;          // S / BLK
constexpr float EPS   = 1e-6f;
constexpr float SCALE = 0.08838834764831845f;  // 1/sqrt(128)
}

typedef __attribute__((ext_vector_type(4)))  unsigned int u32x4;
typedef __attribute__((ext_vector_type(4)))  int          i32x4;
typedef __attribute__((ext_vector_type(8)))  int          i32x8;
typedef __attribute__((ext_vector_type(4)))  float        f32x4;
typedef __attribute__((ext_vector_type(8)))  float        f32x8;
typedef __attribute__((ext_vector_type(16))) __bf16       bf16x16;

union Frag {
  u32x4   u[2];   // two 16B chunks = 16 bf16
  bf16x16 v;
};

__device__ inline unsigned short f2bf(float f) {
  unsigned int u = __builtin_bit_cast(unsigned int, f);
  u += 0x7fffu + ((u >> 16) & 1u);            // round-to-nearest-even
  return (unsigned short)(u >> 16);
}
__device__ inline float bf2f(unsigned short b) {
  return __builtin_bit_cast(float, ((unsigned int)b) << 16);
}

__device__ inline f32x8 wmma_bf16(const bf16x16 a, const bf16x16 b, const f32x8 c) {
  // (neg_a, A, neg_b, B, c_mod, C, reuse_a, reuse_b)
  return __builtin_amdgcn_wmma_f32_16x16x32_bf16(false, a, false, b, (short)0, c,
                                                 false, false);
}

#if defined(__AMDGCN__) && __has_builtin(__builtin_amdgcn_tensor_load_to_lds) && \
    __has_builtin(__builtin_amdgcn_s_wait_tensorcnt)
#define K5_USE_TDM 1
#else
#define K5_USE_TDM 0
#endif

#if K5_USE_TDM
// ---------------------------------------------------------------------------
// TDM 2-D tile load: tile_dim0 = 32 bf16 (64B), tile_rows rows, tensor row
// stride 2048 elems. LDS padding: after every 16 DWORDs (one 32-elem row)
// insert 8 DWORDs  ->  48-element padded LDS row stride (matches LSTR).
// D# bitfields per CDNA5 ISA 8.3 / 8.4 (group0 / group1); groups 2-3 zero.
// ---------------------------------------------------------------------------
__device__ inline void tdm_load_tile(unsigned lds_off,
                                     const unsigned short* gsrc,
                                     int tile_rows) {
  unsigned long long ga = (unsigned long long)(uintptr_t)gsrc;
  u32x4 g0;
  g0[0] = 1u;                                        // count=1, user mode
  g0[1] = lds_off;                                   // lds_addr (bytes)
  g0[2] = (unsigned)(ga & 0xffffffffu);              // global_addr[31:0]
  g0[3] = (unsigned)((ga >> 32) & 0x01ffffffu)       // global_addr[56:32]
        | (2u << 30);                                // type = 2 ("image")
  i32x8 g1;
  g1[0] = (int)((1u << 16)                           // data_size = 2 bytes
              | (1u << 20)                           // pad_enable
              | (3u << 22)                           // pad_interval: 16 DWORDs
              | (7u << 25));                         // pad_amount:   8 DWORDs
  g1[1] = (int)(2048u << 16);                        // tensor_dim0 = 2048 (lo16)
  g1[2] = (int)(2048u << 16);                        // dim0 hi | tensor_dim1 lo16
  g1[3] = (int)(32u << 16);                          // dim1 hi | tile_dim0 = 32
  g1[4] = tile_rows;                                 // tile_dim1 ; tile_dim2 = 0
  g1[5] = 2048;                                      // tensor_dim0_stride lo32
  g1[6] = 0;                                         // stride hi | dim1_stride
  g1[7] = 0;
  i32x4 g2 = {0, 0, 0, 0};
  i32x4 g3 = {0, 0, 0, 0};
#if defined(__clang_major__) && __clang_major__ >= 23
  i32x8 g4 = {0, 0, 0, 0, 0, 0, 0, 0};
  __builtin_amdgcn_tensor_load_to_lds(g0, g1, g2, g3, g4, 0);
#else
  __builtin_amdgcn_tensor_load_to_lds(g0, g1, g2, g3, 0);
#endif
}
#endif  // K5_USE_TDM

// ---------------------------------------------------------------------------
// fp32 -> bf16 bulk convert (8 elems / lane, b128 in / b128 out)
// ---------------------------------------------------------------------------
__global__ __launch_bounds__(256) void k5_cvt_bf16(const float* __restrict__ in,
                                                   unsigned short* __restrict__ out,
                                                   int n) {
  int i = (blockIdx.x * 256 + threadIdx.x) * 8;
  if (i + 8 > n) return;
  f32x4 a = *(const f32x4*)(in + i);
  f32x4 b = *(const f32x4*)(in + i + 4);
  u32x4 o;
  o[0] = (unsigned)f2bf(a[0]) | ((unsigned)f2bf(a[1]) << 16);
  o[1] = (unsigned)f2bf(a[2]) | ((unsigned)f2bf(a[3]) << 16);
  o[2] = (unsigned)f2bf(b[0]) | ((unsigned)f2bf(b[1]) << 16);
  o[3] = (unsigned)f2bf(b[2]) | ((unsigned)f2bf(b[3]) << 16);
  *(u32x4*)(out + i) = o;
}

// ---------------------------------------------------------------------------
// out[M,N] = A[M,K](bf16) * W[N,K]^T(bf16) + bias[N]   (M=N=K=2048)
// 256 thr = 8 waves (2 Mrows x 4 Ncols); workgroup tile 128x128;
// wave tile 64x32 -> 8 WMMA per K-step per wave (24 B ds per WMMA).
// LDS double-buffered, padded row stride 48 elems (96B) -> aligned b128 ds.
// Tiles staged by TDM (async, TENSORcnt) with hardware pad, else manually.
// ---------------------------------------------------------------------------
namespace k5 { constexpr int TM = 128, TN = 128, TK = 32, LSTR = 48; }

__global__ __launch_bounds__(256) void k5_gemm_bias_wmma(
    const unsigned short* __restrict__ A,
    const unsigned short* __restrict__ W,
    const float* __restrict__ bias,
    float* __restrict__ out) {
  using namespace k5;
  __shared__ __align__(16) unsigned short As[2][TM * LSTR];
  __shared__ __align__(16) unsigned short Ws[2][TN * LSTR];

  const int t    = threadIdx.x;
  const int m0   = blockIdx.y * TM;
  const int n0   = blockIdx.x * TN;
  const int wave = t >> 5, lane = t & 31;
  const int wr = wave >> 2, wc = wave & 3;     // 2 x 4 wave grid
  const int lm = lane & 15, lh = lane >> 4;

#if K5_USE_TDM
  const unsigned ldsA[2] = {(unsigned)(uintptr_t)&As[0][0],
                            (unsigned)(uintptr_t)&As[1][0]};
  const unsigned ldsW[2] = {(unsigned)(uintptr_t)&Ws[0][0],
                            (unsigned)(uintptr_t)&Ws[1][0]};
  if (wave == 0) {                       // TDM ignores EXEC; wave-uniform branch
    tdm_load_tile(ldsA[0], A + (size_t)m0 * C, TM);
    tdm_load_tile(ldsW[0], W + (size_t)n0 * C, TN);
    __builtin_amdgcn_s_wait_tensorcnt((short)0);
  }
#else
  auto stage = [&](int buf, int k0) {
#pragma unroll
    for (int i = 0; i < 2; ++i) {              // A tile: 128x32
      int lin = (t + i * 256) * 8;
      int r = lin / TK, c = lin % TK;
      u32x4 d = *(const u32x4*)(A + (size_t)(m0 + r) * C + k0 + c);
      *(u32x4*)(&As[buf][r * LSTR + c]) = d;
    }
#pragma unroll
    for (int i = 0; i < 2; ++i) {              // W tile: 128x32
      int lin = (t + i * 256) * 8;
      int r = lin / TK, c = lin % TK;
      u32x4 d = *(const u32x4*)(W + (size_t)(n0 + r) * C + k0 + c);
      *(u32x4*)(&Ws[buf][r * LSTR + c]) = d;
    }
  };
  stage(0, 0);
#endif

  f32x8 acc[4][2];
#pragma unroll
  for (int mt = 0; mt < 4; ++mt)
#pragma unroll
    for (int nt = 0; nt < 2; ++nt)
#pragma unroll
      for (int r = 0; r < 8; ++r) acc[mt][nt][r] = 0.0f;

  __syncthreads();

  for (int k0 = 0; k0 < C; k0 += TK) {
    const int buf = (k0 / TK) & 1;
    if (k0 + TK < C) {
#if K5_USE_TDM
      if (wave == 0) {
        tdm_load_tile(ldsA[buf ^ 1], A + (size_t)m0 * C + k0 + TK, TM);
        tdm_load_tile(ldsW[buf ^ 1], W + (size_t)n0 * C + k0 + TK, TN);
      }
#else
      stage(buf ^ 1, k0 + TK);
#endif
    }

    Frag a[4], b[2];
#pragma unroll
    for (int mt = 0; mt < 4; ++mt) {           // A frag: lane row = lm, K by lh
      const unsigned short* base = &As[buf][(wr * 64 + mt * 16 + lm) * LSTR];
      a[mt].u[0] = *(const u32x4*)(base + lh * 16);
      a[mt].u[1] = *(const u32x4*)(base + lh * 16 + 8);
    }
#pragma unroll
    for (int nt = 0; nt < 2; ++nt) {           // B frag: lane col = lm, 16 K by lh
      const unsigned short* base = &Ws[buf][(wc * 32 + nt * 16 + lm) * LSTR + lh * 16];
      b[nt].u[0] = *(const u32x4*)(base);
      b[nt].u[1] = *(const u32x4*)(base + 8);
    }
#pragma unroll
    for (int mt = 0; mt < 4; ++mt)
#pragma unroll
      for (int nt = 0; nt < 2; ++nt)
        acc[mt][nt] = wmma_bf16(a[mt].v, b[nt].v, acc[mt][nt]);

#if K5_USE_TDM
    if (wave == 0 && k0 + TK < C)
      __builtin_amdgcn_s_wait_tensorcnt((short)0);
#endif
    __syncthreads();
  }

#pragma unroll
  for (int mt = 0; mt < 4; ++mt)
#pragma unroll
    for (int nt = 0; nt < 2; ++nt) {
      const int col = n0 + wc * 32 + nt * 16 + lm;
      const float bb = bias[col];
#pragma unroll
      for (int r = 0; r < 8; ++r) {            // C layout: M = r + 8*lh, N = lm
        const int row = m0 + wr * 64 + mt * 16 + r + 8 * lh;
        out[(size_t)row * C + col] = acc[mt][nt][r] + bb;
      }
    }
}

// ---------------------------------------------------------------------------
// RMSNorm(q,k) + RoPE; convert to bf16.  q,k -> (H,S,D); v -> transposed (H,D,S)
// one 128-thread block per (s,h)
// ---------------------------------------------------------------------------
__global__ __launch_bounds__(128) void k5_rmsnorm_rope(
    const float* __restrict__ qf, const float* __restrict__ kf,
    const float* __restrict__ vf,
    const float* __restrict__ qn_w, const float* __restrict__ kn_w,
    const float* __restrict__ rope,
    unsigned short* __restrict__ q16, unsigned short* __restrict__ k16,
    unsigned short* __restrict__ vT16) {
  using namespace k5;
  const int bid = blockIdx.x;
  const int s = bid >> 4, h = bid & 15;
  const int d = threadIdx.x;
  const size_t idx = (size_t)s * C + h * HD + d;
  const float qv = qf[idx], kv = kf[idx], vv = vf[idx];

  float sq = qv * qv, sk = kv * kv;
#pragma unroll
  for (int off = 16; off > 0; off >>= 1) {
    sq += __shfl_xor(sq, off);
    sk += __shfl_xor(sk, off);
  }
  __shared__ float rq[4], rk[4];
  const int w = d >> 5, ln = d & 31;
  if (ln == 0) { rq[w] = sq; rk[w] = sk; }
  __syncthreads();
  sq = rq[0] + rq[1] + rq[2] + rq[3];
  sk = rk[0] + rk[1] + rk[2] + rk[3];
  const float rsq = rsqrtf(sq * (1.0f / HD) + EPS);
  const float rsk = rsqrtf(sk * (1.0f / HD) + EPS);
  const float qn = qv * rsq * qn_w[d];
  const float kn = kv * rsk * kn_w[d];

  // rope element (s, i, r, c) at s*256 + i*4 + r*2 + c ; cos=(0,0), sin=(1,0)
  const int i = d >> 1;
  const float cs = rope[(size_t)s * 256 + i * 4 + 0];
  const float sn = rope[(size_t)s * 256 + i * 4 + 2];
  const float qp = __shfl_xor(qn, 1);   // pair partner (even<->odd, same wave)
  const float kp = __shfl_xor(kn, 1);
  float qr, kr;
  if ((d & 1) == 0) { qr = cs * qn - sn * qp; kr = cs * kn - sn * kp; }
  else              { qr = sn * qp + cs * qn; kr = sn * kp + cs * kn; }

  q16[((size_t)h * S + s) * HD + d] = f2bf(qr);
  k16[((size_t)h * S + s) * HD + d] = f2bf(kr);
  vT16[((size_t)h * HD + d) * S + s] = f2bf(vv);
}

// ---------------------------------------------------------------------------
// block means over BLK=64 rows: qa/ka (H, 32, 128)
// ---------------------------------------------------------------------------
__global__ __launch_bounds__(128) void k5_block_means(
    const unsigned short* __restrict__ q16,
    const unsigned short* __restrict__ k16,
    float* __restrict__ qa, float* __restrict__ ka) {
  using namespace k5;
  const int h = blockIdx.x >> 5, b = blockIdx.x & 31;
  const int d = threadIdx.x;
  const unsigned short* src = (blockIdx.y == 0) ? q16 : k16;
  float* dst = (blockIdx.y == 0) ? qa : ka;
  float sum = 0.0f;
  const unsigned short* base = src + ((size_t)h * S + b * BLK) * HD + d;
#pragma unroll 8
  for (int i = 0; i < BLK; ++i) sum += bf2f(base[(size_t)i * HD]);
  dst[((size_t)h * S1 + b) * HD + d] = sum * (1.0f / BLK);
}

// ---------------------------------------------------------------------------
// Nabla mask: softmax over 32 key-blocks, keep smallest-prefix complement
// (cumsum-in-ascending-order >= 1-P), OR with the +-2 band (== sta_mask).
// One wave per (h, qblock); lane j = key block.
// ---------------------------------------------------------------------------
__global__ __launch_bounds__(32) void k5_nabla_mask(
    const float* __restrict__ qa, const float* __restrict__ ka,
    int* __restrict__ bmask) {
  using namespace k5;
  const int h = blockIdx.x >> 5, qb = blockIdx.x & 31;
  const int j = threadIdx.x;

  float dot = 0.0f;
  const float* qv = qa + ((size_t)h * S1 + qb) * HD;
  const float* kv = ka + ((size_t)h * S1 + j) * HD;
#pragma unroll 8
  for (int d = 0; d < HD; ++d) dot += qv[d] * kv[d];
  const float sc = dot * SCALE;

  float mx = sc;
#pragma unroll
  for (int off = 16; off > 0; off >>= 1) mx = fmaxf(mx, __shfl_xor(mx, off));
  const float e = __expf(sc - mx);
  float sum = e;
#pragma unroll
  for (int off = 16; off > 0; off >>= 1) sum += __shfl_xor(sum, off);
  const float p = e / sum;

  // cumulative mass of all entries sorting (stably) at or before lane j
  float cum = 0.0f;
  for (int i = 0; i < 32; ++i) {
    const float pi = __shfl(p, i);
    if (pi < p || (pi == p && i <= j)) cum += pi;
  }
  const bool sel = cum >= (1.0f - 0.9f);
  int dd = qb - j; if (dd < 0) dd = -dd;
  const bool band = dd <= 2;
  bmask[((size_t)h * S1 + qb) * S1 + j] = (sel || band) ? 1 : 0;
}

// ---------------------------------------------------------------------------
// Block-sparse flash attention. 4 waves per (h, qblock); each wave owns 16
// query rows. Per selected key block: 16 WMMA (QK^T) + online softmax (half-
// wave shuffles) + P relayout through wave-private LDS + 16 WMMA (PV).
// Output written as bf16 into attn16 (S, C) for the final projection GEMM.
// ---------------------------------------------------------------------------
__global__ __launch_bounds__(128) void k5_sparse_attn(
    const unsigned short* __restrict__ q16,
    const unsigned short* __restrict__ k16,
    const unsigned short* __restrict__ vT16,
    const int* __restrict__ bmask,
    unsigned short* __restrict__ attn16) {
  using namespace k5;
  const int h = blockIdx.x >> 5, qb = blockIdx.x & 31;
  const int wave = threadIdx.x >> 5, lane = threadIdx.x & 31;
  const int lm = lane & 15, lh = lane >> 4;

  __shared__ __align__(16) unsigned short Pl[4][16][72];  // per-wave P tile

  // Q fragments (16 rows x 128), 4 K-chunks, loaded once
  Frag qfrag[4];
  const unsigned short* qrow = q16 + ((size_t)h * S + qb * BLK + wave * 16 + lm) * HD;
#pragma unroll
  for (int kc = 0; kc < 4; ++kc) {
    qfrag[kc].u[0] = *(const u32x4*)(qrow + kc * 32 + lh * 16);
    qfrag[kc].u[1] = *(const u32x4*)(qrow + kc * 32 + lh * 16 + 8);
  }

  f32x8 oacc[8];
  float rm[8], rl[8];
#pragma unroll
  for (int nt = 0; nt < 8; ++nt)
#pragma unroll
    for (int r = 0; r < 8; ++r) oacc[nt][r] = 0.0f;
#pragma unroll
  for (int r = 0; r < 8; ++r) { rm[r] = -1e30f; rl[r] = 0.0f; }

  const int* mrow = bmask + ((size_t)h * S1 + qb) * S1;

  for (int kb = 0; kb < S1; ++kb) {
    if (!mrow[kb]) continue;                   // uniform branch: EXEC stays full

    // ---- S = Q Kb^T -------------------------------------------------------
    f32x8 sacc[4];
#pragma unroll
    for (int nt = 0; nt < 4; ++nt)
#pragma unroll
      for (int r = 0; r < 8; ++r) sacc[nt][r] = 0.0f;

#pragma unroll
    for (int nt = 0; nt < 4; ++nt) {
      const unsigned short* krow = k16 + ((size_t)h * S + kb * BLK + nt * 16 + lm) * HD;
#pragma unroll
      for (int kc = 0; kc < 4; ++kc) {
        Frag bf;
        bf.u[0] = *(const u32x4*)(krow + kc * 32 + lh * 16);
        bf.u[1] = *(const u32x4*)(krow + kc * 32 + lh * 16 + 8);
        sacc[nt] = wmma_bf16(qfrag[kc].v, bf.v, sacc[nt]);
      }
    }

    // ---- online softmax (row m = r + 8*lh lives on one 16-lane half) ------
#pragma unroll
    for (int r = 0; r < 8; ++r) {
      float mx = -1e30f;
#pragma unroll
      for (int nt = 0; nt < 4; ++nt) {
        sacc[nt][r] *= SCALE;
        mx = fmaxf(mx, sacc[nt][r]);
      }
#pragma unroll
      for (int off = 8; off > 0; off >>= 1) mx = fmaxf(mx, __shfl_xor(mx, off));
      const float nm = fmaxf(rm[r], mx);
      const float alpha = __expf(rm[r] - nm);
      float rs = 0.0f;
#pragma unroll
      for (int nt = 0; nt < 4; ++nt) {
        const float p = __expf(sacc[nt][r] - nm);
        sacc[nt][r] = p;
        rs += p;
      }
#pragma unroll
      for (int off = 8; off > 0; off >>= 1) rs += __shfl_xor(rs, off);
      rl[r] = rl[r] * alpha + rs;
#pragma unroll
      for (int nt = 0; nt < 8; ++nt) oacc[nt][r] *= alpha;
      rm[r] = nm;
    }

    // ---- P -> LDS (C-layout) then reload in A-fragment layout -------------
#pragma unroll
    for (int nt = 0; nt < 4; ++nt)
#pragma unroll
      for (int r = 0; r < 8; ++r)
        Pl[wave][r + 8 * lh][nt * 16 + lm] = f2bf(sacc[nt][r]);

    // ---- O += P Vb  (V stored transposed: (H, D, S), contiguous along s) --
#pragma unroll
    for (int kc = 0; kc < 2; ++kc) {
      Frag pa;
      const unsigned short* prow = &Pl[wave][lm][0];
      pa.u[0] = *(const u32x4*)(prow + kc * 32 + lh * 16);
      pa.u[1] = *(const u32x4*)(prow + kc * 32 + lh * 16 + 8);
#pragma unroll
      for (int nt = 0; nt < 8; ++nt) {
        const unsigned short* vrow =
            vT16 + ((size_t)h * HD + nt * 16 + lm) * S + kb * BLK;
        Frag vb;
        vb.u[0] = *(const u32x4*)(vrow + kc * 32 + lh * 16);
        vb.u[1] = *(const u32x4*)(vrow + kc * 32 + lh * 16 + 8);
        oacc[nt] = wmma_bf16(pa.v, vb.v, oacc[nt]);
      }
    }
  }

  // ---- normalize and emit bf16 rows into (S, C) ---------------------------
#pragma unroll
  for (int r = 0; r < 8; ++r) {
    const float inv = 1.0f / rl[r];            // band guarantees rl > 0
    const int row = qb * BLK + wave * 16 + r + 8 * lh;
#pragma unroll
    for (int nt = 0; nt < 8; ++nt)
      attn16[(size_t)row * C + h * HD + nt * 16 + lm] = f2bf(oacc[nt][r] * inv);
  }
}

// ---------------------------------------------------------------------------
// host launcher
// ---------------------------------------------------------------------------
extern "C" void kernel_launch(void* const* d_in, const int* in_sizes, int n_in,
                              void* d_out, int out_size, void* d_ws, size_t ws_size,
                              hipStream_t stream) {
  using namespace k5;
  (void)in_sizes; (void)n_in; (void)out_size; (void)ws_size;

  const float* hidden = (const float*)d_in[0];
  const float* Wq = (const float*)d_in[1];
  const float* bq = (const float*)d_in[2];
  const float* Wk = (const float*)d_in[3];
  const float* bk = (const float*)d_in[4];
  const float* Wv = (const float*)d_in[5];
  const float* bv = (const float*)d_in[6];
  const float* qn_w = (const float*)d_in[7];
  const float* kn_w = (const float*)d_in[8];
  const float* Wo = (const float*)d_in[9];
  const float* bo = (const float*)d_in[10];
  const float* rope = (const float*)d_in[11];
  // d_in[12] (sta_mask) is the |i-j|<=2 band, recomputed analytically in-kernel.

  char* ws = (char*)d_ws;
  size_t off = 0;
  auto take = [&](size_t bytes) -> char* {
    char* p = ws + off;
    off = (off + bytes + 255) & ~(size_t)255;
    return p;
  };
  const size_t BF_SC = (size_t)S * C * sizeof(unsigned short);
  const size_t F_SC  = (size_t)S * C * sizeof(float);

  unsigned short* hid16  = (unsigned short*)take(BF_SC);
  unsigned short* wq16   = (unsigned short*)take(BF_SC);
  unsigned short* wk16   = (unsigned short*)take(BF_SC);
  unsigned short* wv16   = (unsigned short*)take(BF_SC);
  unsigned short* wo16   = (unsigned short*)take(BF_SC);
  float*          qfp    = (float*)take(F_SC);
  float*          kfp    = (float*)take(F_SC);
  float*          vfp    = (float*)take(F_SC);
  unsigned short* q16    = (unsigned short*)take(BF_SC);
  unsigned short* k16    = (unsigned short*)take(BF_SC);
  unsigned short* vT16   = (unsigned short*)take(BF_SC);
  unsigned short* attn16 = (unsigned short*)take(BF_SC);
  float*          qa     = (float*)take((size_t)NH * S1 * HD * sizeof(float));
  float*          ka     = (float*)take((size_t)NH * S1 * HD * sizeof(float));
  int*            bmask  = (int*)take((size_t)NH * S1 * S1 * sizeof(int));

  const int NEL = S * C;
  const dim3 cvtGrid(NEL / (256 * 8)), cvtBlk(256);
  k5_cvt_bf16<<<cvtGrid, cvtBlk, 0, stream>>>(hidden, hid16, NEL);
  k5_cvt_bf16<<<cvtGrid, cvtBlk, 0, stream>>>(Wq, wq16, NEL);
  k5_cvt_bf16<<<cvtGrid, cvtBlk, 0, stream>>>(Wk, wk16, NEL);
  k5_cvt_bf16<<<cvtGrid, cvtBlk, 0, stream>>>(Wv, wv16, NEL);
  k5_cvt_bf16<<<cvtGrid, cvtBlk, 0, stream>>>(Wo, wo16, NEL);

  const dim3 gemmGrid(C / TN, S / TM), gemmBlk(256);
  k5_gemm_bias_wmma<<<gemmGrid, gemmBlk, 0, stream>>>(hid16, wq16, bq, qfp);
  k5_gemm_bias_wmma<<<gemmGrid, gemmBlk, 0, stream>>>(hid16, wk16, bk, kfp);
  k5_gemm_bias_wmma<<<gemmGrid, gemmBlk, 0, stream>>>(hid16, wv16, bv, vfp);

  k5_rmsnorm_rope<<<dim3(S * NH), dim3(128), 0, stream>>>(
      qfp, kfp, vfp, qn_w, kn_w, rope, q16, k16, vT16);

  k5_block_means<<<dim3(NH * S1, 2), dim3(128), 0, stream>>>(q16, k16, qa, ka);

  k5_nabla_mask<<<dim3(NH * S1), dim3(32), 0, stream>>>(qa, ka, bmask);

  k5_sparse_attn<<<dim3(NH * S1), dim3(128), 0, stream>>>(
      q16, k16, vT16, bmask, attn16);

  k5_gemm_bias_wmma<<<gemmGrid, gemmBlk, 0, stream>>>(
      attn16, wo16, bo, (float*)d_out);
}
